// SetGather_76063870812673
// MI455X (gfx1250) — matrix operand order repaired and checked by reading, not compiled, not run on hardware
//
#include <hip/hip_runtime.h>
#include <math.h>

// Problem constants (match reference)
#define NATOMS   262144
#define BATCH    4096
#define NH       512          // N_HIDDEN
#define K2H      1024         // 2*NH  (GEMM K)
#define N4H      2048         // 4*NH  (GEMM N)
#define M_ITERS  6
#define CHUNK    16           // atoms staged in LDS per step (32 KB)

typedef __attribute__((ext_vector_type(16))) __bf16 v16bf;
typedef __attribute__((ext_vector_type(8)))  float  v8f;

static __device__ __forceinline__ unsigned short f2bf(float f) {
    unsigned int u = __float_as_uint(f);
    u += 0x7fffu + ((u >> 16) & 1u);      // round-to-nearest-even
    return (unsigned short)(u >> 16);
}

// ---------------------------------------------------------------------------
// Segment offsets: offs[b] = first atom index with split >= b  (split sorted)
// ---------------------------------------------------------------------------
__global__ void k_offsets(const long long* __restrict__ split, int* __restrict__ offs) {
    int b = blockIdx.x * blockDim.x + threadIdx.x;
    if (b > BATCH) return;
    int lo = 0, hi = NATOMS;
    while (lo < hi) {
        int mid = (lo + hi) >> 1;
        if (split[mid] < (long long)b) lo = mid + 1; else hi = mid;
    }
    offs[b] = lo;
}

// ---------------------------------------------------------------------------
// Pack U (fp32 row-major [1024][2048]) -> bf16 B-fragment layout.
// B 32x16 (16-bit) VGPR layout: lane 0-15 VGPR0-3 K=0..7, VGPR4-7 K=16..23 (2/VGPR), lane 16-31: +8 on K,
// N = lane%16.  One thread per packed dword (VGPR v holds K=k0,k0+1).
// ---------------------------------------------------------------------------
__global__ void k_upack(const float* __restrict__ U, unsigned int* __restrict__ Ub) {
    int idx  = blockIdx.x * 256 + threadIdx.x;       // 128*32*32*8 = 1M dwords
    int v    = idx & 7;
    int lane = (idx >> 3) & 31;
    int kt   = (idx >> 8) & 31;
    int nt   = idx >> 13;
    int n  = nt * 16 + (lane & 15);
    int k0 = kt * 32 + ((v < 4) ? (2 * v + 8 * (lane >> 4)) : (16 + 2 * (v - 4) + 8 * (lane >> 4)));
    unsigned int lo = f2bf(U[(size_t)k0 * N4H + n]);
    unsigned int hi = f2bf(U[(size_t)(k0 + 1) * N4H + n]);
    Ub[idx] = lo | (hi << 16);
}

// ---------------------------------------------------------------------------
// Fused attention per segment: e = feat.h, online (flash) softmax, r readout.
// One 256-thread block per segment. Atom chunk staged in LDS and reused.
// ---------------------------------------------------------------------------
__global__ __launch_bounds__(256) void k_attn(const float* __restrict__ feat,
                                              const int*   __restrict__ offs,
                                              const float* __restrict__ h,
                                              float*       __restrict__ r) {
    __shared__ float4 hs[NH / 4];             // h[b], 2 KB
    __shared__ float  featL[CHUNK][NH];       // 32 KB staged atom features
    __shared__ float  eL[CHUNK];
    __shared__ float  pL[CHUNK];

    const int b     = blockIdx.x;
    const int t     = threadIdx.x;
    const int lane  = t & 31;
    const int w     = t >> 5;
    const int start = offs[b];
    const int cnt   = offs[b + 1] - start;

    for (int i = t; i < NH / 4; i += 256)
        hs[i] = reinterpret_cast<const float4*>(h + (size_t)b * NH)[i];
    __syncthreads();

    float m_run = -__builtin_inff();
    float l_run = 0.f;
    float r0 = 0.f, r1 = 0.f;

    for (int cs = 0; cs < cnt; cs += CHUNK) {
        const int cc = min(CHUNK, cnt - cs);

        // ---- phase A: wave per atom: global->regs (dot) + regs->LDS (stage)
        for (int li = w; li < cc; li += 8) {
            const int atom = start + cs + li;
            const float4* fp = reinterpret_cast<const float4*>(feat + (size_t)atom * NH) + lane * 4;
            float4 f0 = fp[0], f1 = fp[1], f2 = fp[2], f3 = fp[3];
            float4 h0 = hs[lane * 4 + 0], h1 = hs[lane * 4 + 1];
            float4 h2 = hs[lane * 4 + 2], h3 = hs[lane * 4 + 3];
            float d =  f0.x * h0.x + f0.y * h0.y + f0.z * h0.z + f0.w * h0.w;
            d += f1.x * h1.x + f1.y * h1.y + f1.z * h1.z + f1.w * h1.w;
            d += f2.x * h2.x + f2.y * h2.y + f2.z * h2.z + f2.w * h2.w;
            d += f3.x * h3.x + f3.y * h3.y + f3.z * h3.z + f3.w * h3.w;
            float4* fl = reinterpret_cast<float4*>(&featL[li][lane * 16]);
            fl[0] = f0; fl[1] = f1; fl[2] = f2; fl[3] = f3;
            #pragma unroll
            for (int off = 16; off > 0; off >>= 1) d += __shfl_xor(d, off, 32);
            if (lane == 0) eL[li] = d;
        }
        // speculative prefetch of the next chunk (global_prefetch_b8)
        if (cs + CHUNK < cnt) {
            const char* nf = reinterpret_cast<const char*>(
                feat + (size_t)(start + cs + CHUNK + w) * NH);
            __builtin_prefetch(nf + lane * 64, 0, 0);
        }
        __syncthreads();

        // ---- phase B: online softmax update (computed redundantly per thread)
        float cmax = -__builtin_inff();
        for (int j = 0; j < cc; ++j) cmax = fmaxf(cmax, eL[j]);
        float m_new = fmaxf(m_run, cmax);
        float scale = (m_run == -__builtin_inff()) ? 0.f : __expf(m_run - m_new);
        if (t < cc) pL[t] = __expf(eL[t] - m_new);
        __syncthreads();

        r0 *= scale; r1 *= scale;
        float ls = 0.f;
        for (int j = 0; j < cc; ++j) {
            float p = pL[j];
            ls += p;
            r0 += p * featL[j][2 * t + 0];
            r1 += p * featL[j][2 * t + 1];
        }
        l_run = l_run * scale + ls;
        m_run = m_new;
        __syncthreads();
    }

    float inv = (l_run > 0.f) ? 1.f / l_run : 0.f;
    r[(size_t)b * NH + 2 * t + 0] = r0 * inv;
    r[(size_t)b * NH + 2 * t + 1] = r1 * inv;
}

// ---------------------------------------------------------------------------
// q* fp32 = [h | r]  -> d_out (also the final result after last iteration)
// ---------------------------------------------------------------------------
__global__ void k_qstar(const float* __restrict__ h, const float* __restrict__ r,
                        float* __restrict__ out) {
    int idx = blockIdx.x * 256 + threadIdx.x;          // 4096*1024
    int row = idx >> 10, col = idx & 1023;
    out[idx] = (col < NH) ? h[(size_t)row * NH + col]
                          : r[(size_t)row * NH + (col - NH)];
}

// ---------------------------------------------------------------------------
// Pack q* -> bf16 A-fragment layout (16-bit A 16x32, ISA 7.12.2):
// lane 0-15: M=lane, VGPR0-3 K=0..7, VGPR4-7 K=16..23; lane 16-31: +8 on K.
// ---------------------------------------------------------------------------
__global__ void k_apack(const float* __restrict__ h, const float* __restrict__ r,
                        unsigned int* __restrict__ Ab) {
    int idx  = blockIdx.x * 256 + threadIdx.x;         // 256*32*32*8 = 2M dwords
    int v    = idx & 7;
    int lane = (idx >> 3) & 31;
    int kt   = (idx >> 8) & 31;
    int mt   = idx >> 13;
    int m   = mt * 16 + (lane & 15);
    int hi8 = (lane >> 4) * 8;
    int k0  = kt * 32 + ((v < 4) ? (2 * v + hi8) : (16 + 2 * (v - 4) + hi8));
    float x0 = (k0     < NH) ? h[(size_t)m * NH + k0]       : r[(size_t)m * NH + (k0 - NH)];
    float x1 = (k0 + 1 < NH) ? h[(size_t)m * NH + k0 + 1]   : r[(size_t)m * NH + (k0 + 1 - NH)];
    Ab[idx] = (unsigned int)f2bf(x0) | ((unsigned int)f2bf(x1) << 16);
}

// ---------------------------------------------------------------------------
// z = q* @ U + b  via v_wmma_f32_16x16x32_bf16.
// Block = 8 waves; wave w computes a 16(M) x 64(N) strip; block tile 128x64.
// B tiles (4 KB / k-step) staged global->LDS with double-buffered
// GLOBAL_LOAD_ASYNC_TO_LDS_B128 (ASYNCcnt), so B is fetched once per block
// and the WMMA stream only waits on the one-tile-ahead prefetch.
// ---------------------------------------------------------------------------
__global__ __launch_bounds__(256) void k_gemm(const v16bf* __restrict__ Ab,
                                              const unsigned short* __restrict__ Bb,
                                              const float* __restrict__ bias,
                                              float* __restrict__ z) {
    // [buf][n-frag][lane*16 bf16] : 2 * 4 * 512 shorts = 8 KB
    __shared__ __align__(32) unsigned short Bsh[2][4][512];

    const int t    = threadIdx.x;
    const int lane = t & 31;
    const int w    = t >> 5;
    const int mt   = blockIdx.y * 8 + w;       // 0..255
    const int nb   = blockIdx.x * 4;           // 0..124

    // cooperative async stage of one 4KB B k-tile: 256 threads x 16B
    const int cn   = t >> 6;                   // fragment 0..3
    const int csub = t & 63;                   // 16B chunk within fragment
    const char* Bg = reinterpret_cast<const char*>(Bb);

    auto issueB = [&](int kt, int buf) {
        const char* src = Bg + (((size_t)(nb + cn) * 32 + kt) * 1024) + (size_t)csub * 16;
        unsigned int dst = (unsigned int)(uintptr_t)&Bsh[buf][cn][csub * 8];
        asm volatile("global_load_async_to_lds_b128 %0, %1, off"
                     :: "v"(dst), "v"(src) : "memory");
    };

    issueB(0, 0);

    v8f acc[4] = {};
    for (int kt = 0; kt < 32; ++kt) {
        const int buf = kt & 1;
        if (kt < 31) {
            issueB(kt + 1, buf ^ 1);
            asm volatile("s_wait_asynccnt 0x1" ::: "memory");  // current tile landed
        } else {
            asm volatile("s_wait_asynccnt 0x0" ::: "memory");
        }
        __syncthreads();

        v16bf a  = Ab[((size_t)mt * 32 + kt) * 32 + lane];
        v16bf b0 = *reinterpret_cast<const v16bf*>(&Bsh[buf][0][lane * 16]);
        v16bf b1 = *reinterpret_cast<const v16bf*>(&Bsh[buf][1][lane * 16]);
        v16bf b2 = *reinterpret_cast<const v16bf*>(&Bsh[buf][2][lane * 16]);
        v16bf b3 = *reinterpret_cast<const v16bf*>(&Bsh[buf][3][lane * 16]);
        acc[0] = __builtin_amdgcn_wmma_f32_16x16x32_bf16(false, a, false, b0, (short)0, acc[0], false, false);
        acc[1] = __builtin_amdgcn_wmma_f32_16x16x32_bf16(false, a, false, b1, (short)0, acc[1], false, false);
        acc[2] = __builtin_amdgcn_wmma_f32_16x16x32_bf16(false, a, false, b2, (short)0, acc[2], false, false);
        acc[3] = __builtin_amdgcn_wmma_f32_16x16x32_bf16(false, a, false, b3, (short)0, acc[3], false, false);

        __syncthreads();   // protect buf from next iteration's async overwrite
    }

    // C/D layout: VGPR rr, lane: m = rr + 8*(lane>=16), n = lane%16
    const int mrow = mt * 16 + 8 * (lane >> 4);
    const int nlo  = lane & 15;
    #pragma unroll
    for (int n = 0; n < 4; ++n) {
        int ncol = (nb + n) * 16 + nlo;
        float bv = bias[ncol];
        #pragma unroll
        for (int rr = 0; rr < 8; ++rr)
            z[(size_t)(mrow + rr) * N4H + ncol] = acc[n][rr] + bv;
    }
}

// ---------------------------------------------------------------------------
// LSTM gates: c = sigm(f)*c + sigm(i)*tanh(g); h = sigm(o)*tanh(c)
// ---------------------------------------------------------------------------
__global__ void k_gates(const float* __restrict__ z, float* __restrict__ h,
                        float* __restrict__ c) {
    int idx = blockIdx.x * 256 + threadIdx.x;          // 4096*512
    int row = idx >> 9, col = idx & (NH - 1);
    const float* zr = z + (size_t)row * N4H;
    float zi = zr[col], zf = zr[NH + col], zo = zr[2 * NH + col], zg = zr[3 * NH + col];
    float ig = 1.f / (1.f + __expf(-zi));
    float fg = 1.f / (1.f + __expf(-zf));
    float og = 1.f / (1.f + __expf(-zo));
    float g  = tanhf(zg);
    float cn = fg * c[idx] + ig * g;
    c[idx] = cn;
    h[idx] = og * tanhf(cn);
}

// ---------------------------------------------------------------------------
extern "C" void kernel_launch(void* const* d_in, const int* in_sizes, int n_in,
                              void* d_out, int out_size, void* d_ws, size_t ws_size,
                              hipStream_t stream) {
    const float*     feat  = (const float*)d_in[0];       // [262144,512] fp32
    const long long* split = (const long long*)d_in[1];   // [262144] int64 (sorted)
    const float*     U     = (const float*)d_in[2];       // [1024,2048] fp32
    const float*     bias  = (const float*)d_in[3];       // [2048] fp32
    float*           out   = (float*)d_out;               // [4096,1024] fp32

    // Workspace carve-up (~68 MB total)
    char* ws = (char*)d_ws;
    int*   offs = (int*)ws;                         ws += ((BATCH + 1) * 4 + 255) & ~255;
    float* h    = (float*)ws;                       ws += (size_t)BATCH * NH * 4;     // 8 MB
    float* c    = (float*)ws;                       ws += (size_t)BATCH * NH * 4;     // 8 MB
    float* r    = (float*)ws;                       ws += (size_t)BATCH * NH * 4;     // 8 MB
    float* z    = (float*)ws;                       ws += (size_t)BATCH * N4H * 4;    // 32 MB
    unsigned int* Ab = (unsigned int*)ws;           ws += (size_t)BATCH * K2H * 2;    // 8 MB
    unsigned int* Ub = (unsigned int*)ws;           ws += (size_t)K2H * N4H * 2;      // 4 MB

    hipMemsetAsync(h, 0, (size_t)BATCH * NH * 4, stream);
    hipMemsetAsync(c, 0, (size_t)BATCH * NH * 4, stream);

    k_offsets<<<(BATCH + 256) / 256, 256, 0, stream>>>(split, offs);
    k_upack<<<(K2H * N4H / 2) / 256, 256, 0, stream>>>(U, Ub);   // 4096 blocks

    for (int it = 0; it < M_ITERS; ++it) {
        k_attn<<<BATCH, 256, 0, stream>>>(feat, offs, h, r);
        k_qstar<<<(BATCH * K2H) / 256, 256, 0, stream>>>(h, r, out);   // q* -> d_out
        if (it < M_ITERS - 1) {     // reference returns q* formed BEFORE last LSTM
            k_apack<<<(BATCH * K2H / 2) / 256, 256, 0, stream>>>(h, r, Ab);
            k_gemm<<<dim3(N4H / 64, BATCH / 128), 256, 0, stream>>>(
                (const v16bf*)Ab, (const unsigned short*)Ub, bias, z);
            k_gates<<<(BATCH * NH) / 256, 256, 0, stream>>>(z, h, c);
        }
    }
}
